// AppearanceTrackletEmb_61263413510701
// MI455X (gfx1250) — compile-verified, matrix operand order checked by reading, not run on hardware
//
#include <hip/hip_runtime.h>
#include <hip/hip_bf16.h>
#include <math.h>

// ---------------------------------------------------------------------------
// MI455X (gfx1250) implementation. All matmuls via v_wmma_f32_16x16x32_bf16
// (f32 accumulation). ~385 GFLOP vs ~450MB traffic -> compute-bound; bf16 WMMA
// is the right precision/path. W slabs double-buffered in LDS, A fragments
// double-buffered in registers, so global/ds latency overlaps the WMMA stream.
// ---------------------------------------------------------------------------

typedef __bf16 bf16;
typedef __attribute__((ext_vector_type(16))) __bf16 v16bf;
typedef __attribute__((ext_vector_type(8)))  __bf16 v8bf;
typedef __attribute__((ext_vector_type(8)))  float  v8f;

#define B_    256
#define L_    65
#define CIN_  2053
#define KP0   2080            // CIN padded to multiple of 32
#define C_    512
#define MTOT  (2 * B_ * L_)   // 33280 rows (left batches then right batches)
#define NB    (2 * B_)        // 512 attention batches
#define LP    96              // L padded to multiple of 32 (att K-dim / Vt cols)

__device__ __constant__ const float kScale = 0.044194173824159216f; // 1/sqrt(512)

// ---------------- WMMA fragment helpers (wave32) ---------------------------
// 16-bit A 16x32 layout (ISA 7.12.2): lane row = lane&15; K base = lane<16?0:8;
// elems 0..7 -> K=base..base+7, elems 8..15 -> K=base+16..base+23.
// B fragment from row-major [N,K] uses the same mapping with row = column n.

__device__ __forceinline__ v16bf ld_frag(const bf16* base, int row0, int ld, int k0) {
    int lane = threadIdx.x & 31;
    int r  = row0 + (lane & 15);
    int kb = k0 + ((lane & 16) ? 8 : 0);
    const bf16* p = base + (long)r * ld + kb;
    v8bf lo = *(const v8bf*)p;
    v8bf hi = *(const v8bf*)(p + 16);
    v16bf f;
#pragma unroll
    for (int i = 0; i < 8; ++i) { f[i] = lo[i]; f[i + 8] = hi[i]; }
    return f;
}

__device__ __forceinline__ v16bf ld_frag_rc(const bf16* base, int row0, int ld,
                                            int k0, int nrows) {
    int lane = threadIdx.x & 31;
    int r  = row0 + (lane & 15);
    int kb = k0 + ((lane & 16) ? 8 : 0);
    v16bf f;
    if (r < nrows) {
        const bf16* p = base + (long)r * ld + kb;
        v8bf lo = *(const v8bf*)p;
        v8bf hi = *(const v8bf*)(p + 16);
#pragma unroll
        for (int i = 0; i < 8; ++i) { f[i] = lo[i]; f[i + 8] = hi[i]; }
    } else {
#pragma unroll
        for (int i = 0; i < 16; ++i) f[i] = (bf16)0.0f;
    }
    return f;
}

__device__ __forceinline__ v8f wmma_bf16(v16bf a, v16bf b, v8f c) {
    return __builtin_amdgcn_wmma_f32_16x16x32_bf16(false, a, false, b,
                                                   (short)0, c, false, false);
}

// One 16x16 D tile: MODE 0 -> bias + store Y[M,C]; MODE 1 -> bias + leaky_relu
// + transposed store to Vt[b][c][m] (b = row/65, m = row%65, LP-stride).
__device__ __forceinline__ void store_tile16(v8f acc, int m0, int c0,
                                             const float* bias, bf16* Y,
                                             bf16* Vt, int mode) {
    int lane  = threadIdx.x & 31;
    int c     = c0 + (lane & 15);
    int rbase = m0 + ((lane & 16) ? 8 : 0);
    float bv  = bias[c];
#pragma unroll
    for (int r = 0; r < 8; ++r) {
        int row = rbase + r;
        float v = acc[r] + bv;
        if (mode == 0) {
            Y[(long)row * C_ + c] = (bf16)v;
        } else {
            v = (v < 0.0f) ? 0.01f * v : v;
            int b = row / L_, m = row % L_;
            Vt[((long)b * C_ + c) * LP + m] = (bf16)v;
        }
    }
}

// ---------------- cast kernels ---------------------------------------------
// Input (f32, odd stride 2053 -> unvectorizable) -> padded bf16 [MTOT x 2080].
__global__ void k_cast_x0(const float* dl, const float* dr, bf16* X0) {
    long i = (long)blockIdx.x * blockDim.x + threadIdx.x;
    if (i >= (long)MTOT * KP0) return;
    long r = i / KP0;
    int  k = (int)(i % KP0);
    float v = 0.0f;
    if (k < CIN_) {
        const float* p = (r < (MTOT / 2)) ? dl + r * CIN_
                                          : dr + (r - MTOT / 2) * CIN_;
        v = p[k];
    }
    X0[i] = (bf16)v;
}

__global__ void k_cast_w0(const float* w, bf16* out) {
    long i = (long)blockIdx.x * blockDim.x + threadIdx.x;   // over 512*2080
    if (i >= (long)C_ * KP0) return;
    int  k = (int)(i % KP0);
    long r = i / KP0;
    out[i] = (k < CIN_) ? (bf16)w[r * CIN_ + k] : (bf16)0.0f;
}

__global__ void k_cast(const float* w, bf16* out, long n) {
    long i = (long)blockIdx.x * blockDim.x + threadIdx.x;
    if (i < n) out[i] = (bf16)w[i];
}

// ---------------- unified QKV GEMM: [MTOT x kdim] x W[512 x kdim]^T --------
// grid (MTOT/128, 512/64), block 256 = 8 waves (4 m-groups x 2 n-groups),
// each wave owns a 32x32 output tile (2x2 WMMA tiles -> A,B frags reused 2x).
// Pipeline: W slab (64x32 bf16 = 4KB) double-buffered in LDS; A fragments
// double-buffered in registers. Step s computes on regs/LDS loaded at s-1
// while issuing step s+1's global loads and LDS staging.
__global__ void k_gemm(const bf16* __restrict__ A, int lda, int kdim,
                       const bf16* __restrict__ W, const float* __restrict__ bias,
                       bf16* Y, bf16* Vt, int mode) {
    __shared__ bf16 sW[2][64][32];
    int t    = threadIdx.x;
    int lane = t & 31;
    int w    = t >> 5;
    int wm   = w & 3;             // m-group 0..3
    int wn   = w >> 2;            // n-group 0..1
    int m0   = blockIdx.x * 128 + wm * 32;
    int n0g  = blockIdx.y * 64;
    int nloc = wn * 32;

    // staging map: thread -> one v8bf (16B) of the 64x32 slab
    int srow = t >> 2;            // 0..63 (local n row)
    int scol = (t & 3) * 8;       // 0,8,16,24 (k col)
    const bf16* wsrc = W + (long)(n0g + srow) * kdim + scol;

    v8f zero = {};
    v8f acc[2][2];
#pragma unroll
    for (int i = 0; i < 2; ++i)
#pragma unroll
        for (int j = 0; j < 2; ++j) acc[i][j] = zero;

    int nsteps = kdim / 32;

    // prologue: stage slab 0, preload A fragments for step 0
    *(v8bf*)&sW[0][srow][scol] = *(const v8bf*)wsrc;
    v16bf a0 = ld_frag(A, m0,      lda, 0);
    v16bf a1 = ld_frag(A, m0 + 16, lda, 0);
    __syncthreads();

    for (int s = 0; s < nsteps; ++s) {
        int cur = s & 1;
        v16bf a0n = a0, a1n = a1;
        if (s + 1 < nsteps) {
            // stage next W slab into the other LDS buffer
            *(v8bf*)&sW[cur ^ 1][srow][scol] =
                *(const v8bf*)(wsrc + (long)(s + 1) * 32);
            // preload next A fragments (covered by this step's WMMAs)
            a0n = ld_frag(A, m0,      lda, (s + 1) * 32);
            a1n = ld_frag(A, m0 + 16, lda, (s + 1) * 32);
            __builtin_prefetch(A + (long)(m0 + (lane & 15)) * lda + (s + 2) * 32, 0, 1);
        }
        v16bf b0 = ld_frag(&sW[cur][0][0], nloc,      32, 0);
        v16bf b1 = ld_frag(&sW[cur][0][0], nloc + 16, 32, 0);
        acc[0][0] = wmma_bf16(a0, b0, acc[0][0]);
        acc[0][1] = wmma_bf16(a0, b1, acc[0][1]);
        acc[1][0] = wmma_bf16(a1, b0, acc[1][0]);
        acc[1][1] = wmma_bf16(a1, b1, acc[1][1]);
        a0 = a0n;
        a1 = a1n;
        __syncthreads();
    }

#pragma unroll
    for (int i = 0; i < 2; ++i)
#pragma unroll
        for (int j = 0; j < 2; ++j)
            store_tile16(acc[i][j], m0 + i * 16, n0g + nloc + j * 16,
                         bias, Y, Vt, mode);
}

// ---------------- attention scores: att = Q . K^T  (per batch, K=512) ------
// grid (5, 5, 512), block 32 (one wave per 16x16 tile); Q/K fragments
// double-buffered in registers to keep loads ahead of the WMMAs.
__global__ void k_att(const bf16* Q, const bf16* K, float* attf) {
    int tm = blockIdx.x, tn = blockIdx.y, b = blockIdx.z;
    const bf16* Qb = Q + (long)b * L_ * C_;
    const bf16* Kb = K + (long)b * L_ * C_;
    v8f acc = {};
    v16bf a  = ld_frag_rc(Qb, tm * 16, C_, 0, L_);
    v16bf bb = ld_frag_rc(Kb, tn * 16, C_, 0, L_);
    for (int k0 = 0; k0 < C_; k0 += 32) {
        v16bf an = a, bn = bb;
        if (k0 + 32 < C_) {
            an = ld_frag_rc(Qb, tm * 16, C_, k0 + 32, L_);
            bn = ld_frag_rc(Kb, tn * 16, C_, k0 + 32, L_);
        }
        acc = wmma_bf16(a, bb, acc);
        a = an; bb = bn;
    }
    int lane = threadIdx.x & 31;
    int l0 = tm * 16 + ((lane & 16) ? 8 : 0);
    int m  = tn * 16 + (lane & 15);
    if (m < L_) {
#pragma unroll
        for (int r = 0; r < 8; ++r) {
            int l = l0 + r;
            if (l < L_) attf[((long)b * L_ + l) * L_ + m] = acc[r];
        }
    }
}

// ---------------- softmax over m (scaled), bf16 out zero-padded to LP ------
// grid (NB*L), block 32 (one wave per row; wave32 shuffle reductions).
__global__ void k_softmax(const float* attf, bf16* attb) {
    int bl = blockIdx.x;
    int t  = threadIdx.x;
    const float* row = attf + (long)bl * L_;
    float v[3];
    float mx = -1e30f;
#pragma unroll
    for (int i = 0; i < 3; ++i) {
        int m = t + i * 32;
        v[i] = (m < L_) ? row[m] * kScale : -1e30f;
        mx = fmaxf(mx, v[i]);
    }
#pragma unroll
    for (int off = 16; off >= 1; off >>= 1) mx = fmaxf(mx, __shfl_xor(mx, off, 32));
    float s = 0.0f;
#pragma unroll
    for (int i = 0; i < 3; ++i) {
        int m = t + i * 32;
        v[i] = (m < L_) ? __expf(v[i] - mx) : 0.0f;
        s += v[i];
    }
#pragma unroll
    for (int off = 16; off >= 1; off >>= 1) s += __shfl_xor(s, off, 32);
    float inv = 1.0f / s;
    bf16* orow = attb + (long)bl * LP;
#pragma unroll
    for (int i = 0; i < 3; ++i) {
        int m = t + i * 32;
        if (m < LP) orow[m] = (bf16)((m < L_) ? v[i] * inv : 0.0f);
    }
}

// ---------------- context: X[l,c] = att[l,:] . Vt[c,:] (+residual) ---------
// grid (5, 8, 512), block 32. A = att [65 x LP], B = Vt rows [512 x LP].
__global__ void k_out(const bf16* attb, const bf16* Vt, bf16* X, int resid) {
    int tm = blockIdx.x, n0 = blockIdx.y * 64, b = blockIdx.z;
    const bf16* A  = attb + (long)b * L_ * LP;
    const bf16* Bv = Vt   + (long)b * C_ * LP;
    v8f zero = {};
    v8f acc[4];
#pragma unroll
    for (int j = 0; j < 4; ++j) acc[j] = zero;

    for (int k0 = 0; k0 < LP; k0 += 32) {
        v16bf a = ld_frag_rc(A, tm * 16, LP, k0, L_);
#pragma unroll
        for (int j = 0; j < 4; ++j) {
            v16bf bb = ld_frag(Bv, n0 + j * 16, LP, k0);
            acc[j] = wmma_bf16(a, bb, acc[j]);
        }
    }
    int lane = threadIdx.x & 31;
    int l0 = tm * 16 + ((lane & 16) ? 8 : 0);
#pragma unroll
    for (int j = 0; j < 4; ++j) {
        int c = n0 + j * 16 + (lane & 15);
#pragma unroll
        for (int r = 0; r < 8; ++r) {
            int l = l0 + r;
            if (l < L_) {
                long idx = ((long)b * L_ + l) * C_ + c;
                float v = acc[j][r];
                if (resid) v += (float)X[idx];
                X[idx] = (bf16)v;
            }
        }
    }
}

// ---------------- head: mean over L, concat, fc1+relu, fc2 -----------------
__global__ void k_pool(const bf16* X, float* h) {
    int gb = blockIdx.x, t = threadIdx.x;
#pragma unroll
    for (int i = 0; i < 2; ++i) {
        int c = t + i * 256;
        float s = 0.0f;
        for (int l = 0; l < L_; ++l) s += (float)X[((long)gb * L_ + l) * C_ + c];
        float mean = s * (1.0f / (float)L_);
        if (gb < B_) h[(long)gb * 1024 + c] = mean;
        else         h[(long)(gb - B_) * 1024 + C_ + c] = mean;
    }
}

__global__ void k_fc1(const float* h, const float* W1, const float* b1, float* h1) {
    int row = blockIdx.x, c = threadIdx.x;
    const float* hr = h  + (long)row * 1024;
    const float* wr = W1 + (long)c * 1024;
    float s = b1[c];
    for (int k = 0; k < 1024; ++k) s += hr[k] * wr[k];
    h1[(long)row * 256 + c] = (s > 0.0f) ? s : 0.0f;
}

__global__ void k_fc2(const float* h1, const float* W2, const float* b2, float* y) {
    int row = blockIdx.x, t = threadIdx.x;
    if (t >= 2) return;
    const float* hr = h1 + (long)row * 256;
    const float* wr = W2 + (long)t * 256;
    float s = b2[t];
    for (int k = 0; k < 256; ++k) s += hr[k] * wr[k];
    y[row * 2 + t] = s;
}

// ---------------------------------------------------------------------------
extern "C" void kernel_launch(void* const* d_in, const int* in_sizes, int n_in,
                              void* d_out, int out_size, void* d_ws, size_t ws_size,
                              hipStream_t stream) {
    const float* dl   = (const float*)d_in[0];
    const float* dr   = (const float*)d_in[1];
    const float* Wq0f = (const float*)d_in[2];
    const float* bq0  = (const float*)d_in[3];
    const float* Wk0f = (const float*)d_in[4];
    const float* bk0  = (const float*)d_in[5];
    const float* Wv0f = (const float*)d_in[6];
    const float* bv0  = (const float*)d_in[7];
    const float* Wqf  = (const float*)d_in[8];
    const float* bqf  = (const float*)d_in[9];
    const float* Wkf  = (const float*)d_in[10];
    const float* bkf  = (const float*)d_in[11];
    const float* Wvf  = (const float*)d_in[12];
    const float* bvf  = (const float*)d_in[13];
    const float* W1f  = (const float*)d_in[14];
    const float* b1f  = (const float*)d_in[15];
    const float* W2f  = (const float*)d_in[16];
    const float* b2f  = (const float*)d_in[17];

    char* p = (char*)d_ws;
    auto alloc = [&](size_t bytes) -> void* {
        void* r = (void*)p;
        p += (bytes + 255) & ~(size_t)255;
        return r;
    };
    bf16*  X0   = (bf16*)alloc((size_t)MTOT * KP0 * 2);   // padded bf16 input
    bf16*  W0q  = (bf16*)alloc((size_t)C_ * KP0 * 2);
    bf16*  W0k  = (bf16*)alloc((size_t)C_ * KP0 * 2);
    bf16*  W0v  = (bf16*)alloc((size_t)C_ * KP0 * 2);
    bf16*  Wqb  = (bf16*)alloc((size_t)3 * C_ * C_ * 2);
    bf16*  Wkb  = (bf16*)alloc((size_t)3 * C_ * C_ * 2);
    bf16*  Wvb  = (bf16*)alloc((size_t)3 * C_ * C_ * 2);
    bf16*  Qb   = (bf16*)alloc((size_t)MTOT * C_ * 2);
    bf16*  Kb   = (bf16*)alloc((size_t)MTOT * C_ * 2);
    bf16*  Vt   = (bf16*)alloc((size_t)NB * C_ * LP * 2);
    float* attf = (float*)alloc((size_t)NB * L_ * L_ * 4);
    bf16*  attb = (bf16*)alloc((size_t)NB * L_ * LP * 2);
    bf16*  X    = (bf16*)alloc((size_t)MTOT * C_ * 2);
    float* h    = (float*)alloc((size_t)B_ * 1024 * 4);
    float* h1   = (float*)alloc((size_t)B_ * 256 * 4);

    // casts (bf16, zero-padded K)
    long nx0 = (long)MTOT * KP0;
    k_cast_x0<<<(int)((nx0 + 255) / 256), 256, 0, stream>>>(dl, dr, X0);
    k_cast_w0<<<(C_ * KP0 + 255) / 256, 256, 0, stream>>>(Wq0f, W0q);
    k_cast_w0<<<(C_ * KP0 + 255) / 256, 256, 0, stream>>>(Wk0f, W0k);
    k_cast_w0<<<(C_ * KP0 + 255) / 256, 256, 0, stream>>>(Wv0f, W0v);
    long nw = 3L * C_ * C_;
    k_cast<<<(int)((nw + 255) / 256), 256, 0, stream>>>(Wqf, Wqb, nw);
    k_cast<<<(int)((nw + 255) / 256), 256, 0, stream>>>(Wkf, Wkb, nw);
    k_cast<<<(int)((nw + 255) / 256), 256, 0, stream>>>(Wvf, Wvb, nw);

    dim3 gG(MTOT / 128, C_ / 64);       // (260, 8)
    dim3 gA(5, 5, NB);
    dim3 gO(5, C_ / 64, NB);

    // ---- layer 0 (CIN -> C, no residual) ----
    k_gemm<<<gG, 256, 0, stream>>>(X0, KP0, KP0, W0q, bq0, Qb, nullptr, 0);
    k_gemm<<<gG, 256, 0, stream>>>(X0, KP0, KP0, W0k, bk0, Kb, nullptr, 0);
    k_gemm<<<gG, 256, 0, stream>>>(X0, KP0, KP0, W0v, bv0, nullptr, Vt, 1);
    k_att<<<gA, 32, 0, stream>>>(Qb, Kb, attf);
    k_softmax<<<NB * L_, 32, 0, stream>>>(attf, attb);
    k_out<<<gO, 32, 0, stream>>>(attb, Vt, X, 0);

    // ---- layers 1..3 (C -> C, residual) ----
    for (int n = 0; n < 3; ++n) {
        const bf16* wq = Wqb + (long)n * C_ * C_;
        const bf16* wk = Wkb + (long)n * C_ * C_;
        const bf16* wv = Wvb + (long)n * C_ * C_;
        const float* bq = bqf + n * C_;
        const float* bk = bkf + n * C_;
        const float* bv = bvf + n * C_;
        k_gemm<<<gG, 256, 0, stream>>>(X, C_, C_, wq, bq, Qb, nullptr, 0);
        k_gemm<<<gG, 256, 0, stream>>>(X, C_, C_, wk, bk, Kb, nullptr, 0);
        k_gemm<<<gG, 256, 0, stream>>>(X, C_, C_, wv, bv, nullptr, Vt, 1);
        k_att<<<gA, 32, 0, stream>>>(Qb, Kb, attf);
        k_softmax<<<NB * L_, 32, 0, stream>>>(attf, attb);
        k_out<<<gO, 32, 0, stream>>>(attb, Vt, X, 1);
    }

    // ---- head ----
    k_pool<<<NB, 256, 0, stream>>>(X, h);
    k_fc1<<<B_, 256, 0, stream>>>(h, W1f, b1f, h1);
    k_fc2<<<B_, 32, 0, stream>>>(h1, W2f, b2f, (float*)d_out);
}
